// GraphClassificationModel_22892175688109
// MI455X (gfx1250) — matrix operand order repaired and checked by reading, not compiled, not run on hardware
//
#include <hip/hip_runtime.h>
#include <hip/hip_bf16.h>

// ---------------------------------------------------------------------------
// GATv2 GNN forward for MI455X (gfx1250, wave32).
// Heavy GEMMs (node transforms + final FC) use V_WMMA_F32_16X16X32_BF16.
// Fragments are loaded straight from global memory (row-major [row,k] layout
// is per-lane contiguous for both A and B^T fragments), so no LDS staging,
// no barriers: pure global_load_b128 -> v_wmma stream, 8 WMMAs per K-step.
// Irregular graph ops use f32 atomics (L2-resident).
// ---------------------------------------------------------------------------

typedef __attribute__((ext_vector_type(16))) __bf16 v16bf;
typedef __attribute__((ext_vector_type(8)))  __bf16 v8bf;
typedef __attribute__((ext_vector_type(8)))  float  v8f;

#define NNODES   10000
#define NGRAPHS  64
#define DMAX     2048
#define NEGSLOPE 0.2f

// ---------------------------------------------------------------------------
// f32 -> bf16 convert with zero-padding of the K dimension to Kpad
// ---------------------------------------------------------------------------
__global__ void k_to_bf16_pad(const float* __restrict__ in, __bf16* __restrict__ out,
                              int rows, int din, int Kpad) {
  size_t tid = (size_t)blockIdx.x * blockDim.x + threadIdx.x;
  size_t total = (size_t)rows * Kpad;
  if (tid >= total) return;
  int r = (int)(tid / Kpad);
  int k = (int)(tid % Kpad);
  out[tid] = (k < din) ? (__bf16)in[(size_t)r * din + k] : (__bf16)0.0f;
}

// ---------------------------------------------------------------------------
// WMMA fragment builders (direct from global memory).
//
// A fragment (16x32 bf16, M=rows of A):
//   lane l (<16):  row m=l,    K = 0..15
//   lane l (>=16): row m=l-16, K = 16..31
//   -> two 16B contiguous chunks at k-offsets half*16 and half*16+8.
//
// B fragment (32x16 bf16, N=rows of W since B = W^T):
//   lane l (<16):  col n=l,    K = 0..15
//   lane l (>=16): col n=l-16, K = 16..31
//   -> two 16B contiguous chunks at k-offsets half*16 and half*16+8.
// ---------------------------------------------------------------------------
__device__ __forceinline__ v16bf frag_a(const __bf16* p) {
  v8bf lo = *(const v8bf*)(p);
  v8bf hi = *(const v8bf*)(p + 8);
  v16bf a;
  #pragma unroll
  for (int i = 0; i < 8; ++i) { a[i] = lo[i]; a[i + 8] = hi[i]; }
  return a;
}
__device__ __forceinline__ v16bf frag_b(const __bf16* p) {
  v8bf lo = *(const v8bf*)(p);
  v8bf hi = *(const v8bf*)(p + 8);
  v16bf b;
  #pragma unroll
  for (int i = 0; i < 8; ++i) { b[i] = lo[i]; b[i + 8] = hi[i]; }
  return b;
}

// ---------------------------------------------------------------------------
// Dual-output bf16 WMMA GEMM:
//   out0[m,n] = sum_k A[m,k]*W0[n,k] + b0[n]   (likewise out1/W1/b1)
// One wave per block; register tile = 16(M) x 64(N) for BOTH weight matrices:
// 8 accumulators, A fragment reused across 8 WMMAs per 32-wide K step.
// ---------------------------------------------------------------------------
__global__ __launch_bounds__(32)
void k_gemm_bf16_dual(const __bf16* __restrict__ A,    // [M, Kpad]
                      const __bf16* __restrict__ W0,   // [Nout, Kpad]
                      const __bf16* __restrict__ W1,   // [Nout, Kpad] or null
                      const float*  __restrict__ b0,
                      const float*  __restrict__ b1,
                      float* __restrict__ out0,        // [M, Nout]
                      float* __restrict__ out1,
                      int Nout, int Kpad) {
  const int lane = threadIdx.x;
  const int half = lane >> 4;          // 0 or 1
  const int r16  = lane & 15;
  const int m0 = blockIdx.x * 16;
  const int n0 = blockIdx.y * 64;
  const bool dual = (W1 != nullptr);

  v8f c0[4] = {{}, {}, {}, {}};
  v8f c1[4] = {{}, {}, {}, {}};

  // per-lane fragment base pointers (16B aligned: Kpad is a multiple of 32)
  const __bf16* pa = A + (size_t)(m0 + r16) * Kpad + half * 16;
  const __bf16* pb0[4];
  const __bf16* pb1[4];
  #pragma unroll
  for (int t = 0; t < 4; ++t) {
    pb0[t] = W0 + (size_t)(n0 + t * 16 + r16) * Kpad + half * 16;
    pb1[t] = dual ? (W1 + (size_t)(n0 + t * 16 + r16) * Kpad + half * 16) : pb0[t];
  }

  if (dual) {
    for (int kt = 0; kt < Kpad; kt += 32) {
      __builtin_prefetch(pa + kt + 32, 0, 3);
      v16bf a = frag_a(pa + kt);
      #pragma unroll
      for (int t = 0; t < 4; ++t) {
        v16bf b = frag_b(pb0[t] + kt);
        c0[t] = __builtin_amdgcn_wmma_f32_16x16x32_bf16(false, a, false, b,
                                                        (short)0, c0[t], false, false);
        v16bf bb = frag_b(pb1[t] + kt);
        c1[t] = __builtin_amdgcn_wmma_f32_16x16x32_bf16(false, a, false, bb,
                                                        (short)0, c1[t], false, false);
      }
    }
  } else {
    for (int kt = 0; kt < Kpad; kt += 32) {
      __builtin_prefetch(pa + kt + 32, 0, 3);
      v16bf a = frag_a(pa + kt);
      #pragma unroll
      for (int t = 0; t < 4; ++t) {
        v16bf b = frag_b(pb0[t] + kt);
        c0[t] = __builtin_amdgcn_wmma_f32_16x16x32_bf16(false, a, false, b,
                                                        (short)0, c0[t], false, false);
      }
    }
  }

  // C/D layout: lanes 0-15 -> rows 0..7 (vgpr j = row j), lanes 16-31 -> rows 8..15
  #pragma unroll
  for (int t = 0; t < 4; ++t) {
    int col = n0 + t * 16 + r16;
    float bv0 = b0[col];
    float bv1 = dual ? b1[col] : 0.0f;
    #pragma unroll
    for (int j = 0; j < 8; ++j) {
      int row = m0 + half * 8 + j;
      out0[(size_t)row * Nout + col] = c0[t][j] + bv0;
      if (dual) out1[(size_t)row * Nout + col] = c1[t][j] + bv1;
    }
  }
}

// ---------------------------------------------------------------------------
// Self-loop attr: scatter-mean of edge_attr by dst
// ---------------------------------------------------------------------------
__global__ void k_edge_count_sum(const float* __restrict__ edge_attr,
                                 const int* __restrict__ dst,
                                 float* __restrict__ cnt, float* __restrict__ ssum, int E) {
  int tid = blockIdx.x * blockDim.x + threadIdx.x;
  if (tid >= E) return;
  int d = dst[tid];
  atomicAdd(&cnt[d], 1.0f);
  #pragma unroll
  for (int c = 0; c < 3; ++c)
    atomicAdd(&ssum[(size_t)d * 3 + c], edge_attr[(size_t)tid * 3 + c]);
}

__global__ void k_finalize_self(const float* __restrict__ ssum, const float* __restrict__ cnt,
                                float* __restrict__ self_attr, int N) {
  int tid = blockIdx.x * blockDim.x + threadIdx.x;
  if (tid >= N) return;
  float inv = 1.0f / fmaxf(cnt[tid], 1.0f);
  #pragma unroll
  for (int c = 0; c < 3; ++c)
    self_attr[(size_t)tid * 3 + c] = ssum[(size_t)tid * 3 + c] * inv;
}

// ---------------------------------------------------------------------------
// Fused per-edge score: score[e] = leaky_relu(xl[s]+xr[d]+attr@We^T) . att
// One wave per edge; lanes stride channels; shuffle reduction.
// ---------------------------------------------------------------------------
__global__ __launch_bounds__(256)
void k_score(const float* __restrict__ xl, const float* __restrict__ xr,
             const float* __restrict__ edge_attr, const float* __restrict__ self_attr,
             const int* __restrict__ src, const int* __restrict__ dst,
             const float* __restrict__ We, const float* __restrict__ att,
             float* __restrict__ score, int E, int Ea, int dout) {
  int wave = blockIdx.x * (blockDim.x >> 5) + (threadIdx.x >> 5);
  int lane = threadIdx.x & 31;
  if (wave >= Ea) return;
  int s, d; const float* ap;
  if (wave < E) { s = src[wave]; d = dst[wave]; ap = edge_attr + (size_t)wave * 3; }
  else          { s = wave - E;  d = s;         ap = self_attr + (size_t)s * 3; }
  float a0 = ap[0], a1 = ap[1], a2 = ap[2];
  const float* xls = xl + (size_t)s * dout;
  const float* xrd = xr + (size_t)d * dout;
  float acc = 0.0f;
  for (int o = lane; o < dout; o += 32) {
    float e = We[o * 3 + 0] * a0 + We[o * 3 + 1] * a1 + We[o * 3 + 2] * a2;
    float m = xls[o] + xrd[o] + e;
    m = (m > 0.0f) ? m : NEGSLOPE * m;
    acc += m * att[o];
  }
  #pragma unroll
  for (int off = 16; off > 0; off >>= 1) acc += __shfl_down(acc, off, 32);
  if (lane == 0) score[wave] = acc;
}

__global__ void k_init_softmax(float* __restrict__ smax, float* __restrict__ denom, int N) {
  int tid = blockIdx.x * blockDim.x + threadIdx.x;
  if (tid >= N) return;
  smax[tid] = -3.0e38f;
  denom[tid] = 0.0f;
}

// float segment-max via order-preserving int/uint atomic trick
__global__ void k_scatter_max(const float* __restrict__ score, const int* __restrict__ dst,
                              float* __restrict__ smax, int E, int Ea) {
  int tid = blockIdx.x * blockDim.x + threadIdx.x;
  if (tid >= Ea) return;
  int d = (tid < E) ? dst[tid] : (tid - E);
  float v = score[tid];
  if (v >= 0.0f) atomicMax((int*)&smax[d], __float_as_int(v));
  else           atomicMin((unsigned int*)&smax[d], __float_as_uint(v));
}

__global__ void k_exp_denom(const float* __restrict__ score, const int* __restrict__ dst,
                            const float* __restrict__ smax, float* __restrict__ aexp,
                            float* __restrict__ denom, int E, int Ea) {
  int tid = blockIdx.x * blockDim.x + threadIdx.x;
  if (tid >= Ea) return;
  int d = (tid < E) ? dst[tid] : (tid - E);
  float v = __expf(score[tid] - smax[d]);
  aexp[tid] = v;
  atomicAdd(&denom[d], v);
}

// h_next[d,:] += (a/denom[d]) * xl[s,:]   (one block per edge)
__global__ __launch_bounds__(256)
void k_aggregate(const float* __restrict__ xl, const float* __restrict__ aexp,
                 const float* __restrict__ denom, const int* __restrict__ src,
                 const int* __restrict__ dst, float* __restrict__ hn,
                 int E, int Ea, int dout) {
  int ea = blockIdx.x;
  if (ea >= Ea) return;
  int s, d;
  if (ea < E) { s = src[ea]; d = dst[ea]; } else { s = ea - E; d = s; }
  float w = aexp[ea] / denom[d];
  const float* xs = xl + (size_t)s * dout;
  float* hd = hn + (size_t)d * dout;
  for (int o = threadIdx.x; o < dout; o += blockDim.x)
    atomicAdd(&hd[o], w * xs[o]);
}

__global__ void k_bias_act(float* __restrict__ h, const float* __restrict__ bias,
                           size_t total, int dout, int do_relu) {
  size_t tid = (size_t)blockIdx.x * blockDim.x + threadIdx.x;
  if (tid >= total) return;
  float v = h[tid] + bias[tid % dout];
  h[tid] = do_relu ? fmaxf(v, 0.0f) : v;
}

__global__ void k_pool(const float* __restrict__ h, const int* __restrict__ batch,
                       float* __restrict__ pooled, int N, int dout) {
  size_t tid = (size_t)blockIdx.x * blockDim.x + threadIdx.x;
  size_t total = (size_t)N * dout;
  if (tid >= total) return;
  int n = (int)(tid / dout);
  int o = (int)(tid % dout);
  atomicAdd(&pooled[(size_t)batch[n] * dout + o], h[tid]);
}

// ---------------------------------------------------------------------------
// Host orchestration
// ---------------------------------------------------------------------------
extern "C" void kernel_launch(void* const* d_in, const int* in_sizes, int n_in,
                              void* d_out, int out_size, void* d_ws, size_t ws_size,
                              hipStream_t stream) {
  const int N  = in_sizes[0] / 3;       // 10000
  const int E  = in_sizes[1] / 2;       // 160000
  const int Ea = E + N;                 // with self loops
  static const int DIMS[7] = {3, 64, 128, 256, 512, 1024, 2048};

  const float* x          = (const float*)d_in[0];
  const int*   edge_index = (const int*)  d_in[1];
  const float* edge_attr  = (const float*)d_in[2];
  const int*   batch      = (const int*)  d_in[3];
  // params flattened in insertion order: 6 x {Wl, bl, Wr, br, We, att, bias}, fc_w, fc_b
  const float* fc_w = (const float*)d_in[4 + 42];
  const float* fc_b = (const float*)d_in[4 + 43];
  const int* src = edge_index;
  const int* dst = edge_index + E;

  // workspace carve-up (256B aligned)
  char* wp = (char*)d_ws;
  auto carve = [&](size_t bytes) -> void* {
    void* p = (void*)wp;
    wp += (bytes + 255) & ~(size_t)255;
    return p;
  };
  float*  hA      = (float*) carve((size_t)N * DMAX * 4);
  float*  hB      = (float*) carve((size_t)N * DMAX * 4);
  float*  xl      = (float*) carve((size_t)N * DMAX * 4);
  float*  xr      = (float*) carve((size_t)N * DMAX * 4);
  __bf16* hb      = (__bf16*)carve((size_t)N * 1024 * 2);
  __bf16* w0b     = (__bf16*)carve((size_t)2048 * 2048 * 2);  // also holds fc_w bf16
  __bf16* w1b     = (__bf16*)carve((size_t)2048 * 1024 * 2);
  float*  score   = (float*) carve((size_t)Ea * 4);
  float*  aexp    = (float*) carve((size_t)Ea * 4);
  float*  smax    = (float*) carve((size_t)N * 4);
  float*  denom   = (float*) carve((size_t)N * 4);
  float*  cnt     = (float*) carve((size_t)N * 4);
  float*  ssum    = (float*) carve((size_t)N * 3 * 4);
  float*  selfa   = (float*) carve((size_t)N * 3 * 4);
  float*  pooled  = (float*) carve((size_t)NGRAPHS * DMAX * 4);
  __bf16* pooledb = (__bf16*)carve((size_t)NGRAPHS * DMAX * 2);

  // --- self-loop edge_attr (scatter-mean by dst) ---
  hipMemsetAsync(cnt,  0, (size_t)N * 4, stream);
  hipMemsetAsync(ssum, 0, (size_t)N * 3 * 4, stream);
  k_edge_count_sum<<<(E + 255) / 256, 256, 0, stream>>>(edge_attr, dst, cnt, ssum, E);
  k_finalize_self<<<(N + 255) / 256, 256, 0, stream>>>(ssum, cnt, selfa, N);

  const float* hcur = x;   // layer-0 input
  float* hbufs[2] = {hA, hB};
  int flip = 0;

  for (int i = 0; i < 6; ++i) {
    const int din  = DIMS[i];
    const int dout = DIMS[i + 1];
    const int Kpad = (din < 32) ? 32 : din;

    const float* Wl   = (const float*)d_in[4 + 7 * i + 0];
    const float* bl   = (const float*)d_in[4 + 7 * i + 1];
    const float* Wr   = (const float*)d_in[4 + 7 * i + 2];
    const float* br   = (const float*)d_in[4 + 7 * i + 3];
    const float* We   = (const float*)d_in[4 + 7 * i + 4];
    const float* att  = (const float*)d_in[4 + 7 * i + 5];
    const float* bias = (const float*)d_in[4 + 7 * i + 6];

    // bf16 conversions (K padded to multiple of 32)
    {
      size_t t;
      t = (size_t)N * Kpad;
      k_to_bf16_pad<<<(unsigned)((t + 255) / 256), 256, 0, stream>>>(hcur, hb, N, din, Kpad);
      t = (size_t)dout * Kpad;
      k_to_bf16_pad<<<(unsigned)((t + 255) / 256), 256, 0, stream>>>(Wl, w0b, dout, din, Kpad);
      k_to_bf16_pad<<<(unsigned)((t + 255) / 256), 256, 0, stream>>>(Wr, w1b, dout, din, Kpad);
    }

    // xl = h@Wl^T + bl ; xr = h@Wr^T + br  (fused dual WMMA GEMM, 16x64 tiles)
    k_gemm_bf16_dual<<<dim3(N / 16, dout / 64), 32, 0, stream>>>(
        hb, w0b, w1b, bl, br, xl, xr, dout, Kpad);

    // per-edge attention scores (fused edge-attr transform)
    k_score<<<(Ea + 7) / 8, 256, 0, stream>>>(xl, xr, edge_attr, selfa, src, dst,
                                              We, att, score, E, Ea, dout);

    // segment softmax over dst
    k_init_softmax<<<(N + 255) / 256, 256, 0, stream>>>(smax, denom, N);
    k_scatter_max<<<(Ea + 255) / 256, 256, 0, stream>>>(score, dst, smax, E, Ea);
    k_exp_denom<<<(Ea + 255) / 256, 256, 0, stream>>>(score, dst, smax, aexp, denom, E, Ea);

    // weighted scatter aggregation
    float* hnext = hbufs[flip];
    hipMemsetAsync(hnext, 0, (size_t)N * dout * 4, stream);
    k_aggregate<<<Ea, 256, 0, stream>>>(xl, aexp, denom, src, dst, hnext, E, Ea, dout);

    size_t tot = (size_t)N * dout;
    k_bias_act<<<(unsigned)((tot + 255) / 256), 256, 0, stream>>>(
        hnext, bias, tot, dout, (i < 5) ? 1 : 0);

    hcur = hnext;
    flip ^= 1;
  }

  // --- global add pool ---
  hipMemsetAsync(pooled, 0, (size_t)NGRAPHS * DMAX * 4, stream);
  {
    size_t tot = (size_t)N * DMAX;
    k_pool<<<(unsigned)((tot + 255) / 256), 256, 0, stream>>>(hcur, batch, pooled, N, DMAX);
  }

  // --- final FC: out = pooled @ fc_w^T + fc_b  (WMMA, single-weight path) ---
  {
    size_t t = (size_t)NGRAPHS * DMAX;
    k_to_bf16_pad<<<(unsigned)((t + 255) / 256), 256, 0, stream>>>(pooled, pooledb, NGRAPHS, DMAX, DMAX);
    t = (size_t)DMAX * DMAX;
    k_to_bf16_pad<<<(unsigned)((t + 255) / 256), 256, 0, stream>>>(fc_w, w0b, DMAX, DMAX, DMAX);
    k_gemm_bf16_dual<<<dim3(NGRAPHS / 16, DMAX / 64), 32, 0, stream>>>(
        pooledb, w0b, nullptr, fc_b, nullptr, (float*)d_out, nullptr, DMAX, DMAX);
  }
}
